// LSTMCell_20005957664971
// MI455X (gfx1250) — matrix operand order repaired and checked by reading, not compiled, not run on hardware
//
#include <hip/hip_runtime.h>
#include <math.h>

#define B_   64
#define T_   256
#define F_   64
#define H_   128
#define H4_  512
#define KPAD 160   // 129 padded to 5 chunks of 32
#define NKC  5
#define NT   32    // 512 gate outputs / 16-wide N tiles

typedef __attribute__((ext_vector_type(16))) __bf16        v16bf;
typedef __attribute__((ext_vector_type(8)))  float         v8f;
typedef __attribute__((ext_vector_type(4)))  unsigned int  v4u;

__device__ __forceinline__ unsigned short f2bf(float f) {
    unsigned int u = __float_as_uint(f);
    unsigned int r = (u + 0x7FFFu + ((u >> 16) & 1u)) >> 16;  // RNE
    return (unsigned short)r;
}
__device__ __forceinline__ float sigmoidf_(float x) { return 1.0f / (1.0f + __expf(-x)); }

// ---------------------------------------------------------------------------
// Pack W_lin (F,129,512) f32 -> bf16, arranged as WMMA B fragments:
// offset = ((f*NT + nt)*NKC + kc)*512 + lane*16 + e
//   n = nt*16 + (lane&15);  k = kc*32 + (lane>=16 ? 8 : 0) + (e&7) + (e>=8 ? 16 : 0);  (k>=129 -> 0)
// ---------------------------------------------------------------------------
__global__ void pack_weights(const float* __restrict__ W_lin,
                             unsigned short* __restrict__ packW) {
    int idx = blockIdx.x * blockDim.x + threadIdx.x;
    const int total = F_ * NT * NKC * 512;
    if (idx >= total) return;
    int e    = idx & 15;
    int lane = (idx >> 4) & 31;
    int rest = idx >> 9;
    int kc   = rest % NKC;  rest /= NKC;
    int nt   = rest % NT;
    int f    = rest / NT;
    int n = nt * 16 + (lane & 15);
    int k = kc * 32 + ((lane >= 16) ? 8 : 0) + (e & 7) + ((e >= 8) ? 16 : 0);
    float v = (k < 129) ? W_lin[((size_t)f * 129 + k) * 512 + n] : 0.0f;
    packW[idx] = f2bf(v);
}

// ---------------------------------------------------------------------------
// One workgroup per sample: sequential scan, WMMA-bf16 GEMV per step.
// A fragment row-0 masking is done by ADDRESS selection: lanes that are not
// matrix row 0 read from a permanently-zeroed LDS region, so no per-chunk
// cndmask/v_nop chain sits in front of the WMMAs.
// ---------------------------------------------------------------------------
__global__ __launch_bounds__(512)
void seq_scan(const float* __restrict__ X, const int* __restrict__ lengths,
              const unsigned short* __restrict__ packW,
              const float* __restrict__ b_lin,
              const float* __restrict__ W_dec, const float* __restrict__ b_dec,
              const float* __restrict__ W_out, const float* __restrict__ b_out,
              float* __restrict__ out) {
    __shared__ float sh_h[F_ * H_];          // 32 KB  : per-feature hidden rows
    __shared__ float sh_gates[H4_];          //  2 KB  : gate pre-activations
    __shared__ float sh_r0[512];
    __shared__ float sh_r1[512];
    // [0] = live bf16 inp_vec, [1] = permanent zeros (A-fragment dead lanes)
    __shared__ alignas(16) unsigned short sh_inpz[2][192];

    const int b    = blockIdx.x;
    const int tid  = threadIdx.x;
    const int lane = tid & 31;
    const int wave = tid >> 5;

    const float* Xt = X + (size_t)b * 4 * T_;
    const float* Xm = Xt + T_;
    const float* Xx = Xt + 2 * T_;
    const float* Xd = Xt + 3 * T_;
    const int len = lengths[b];

    for (int i = tid; i < F_ * H_; i += 512) sh_h[i] = 0.0f;
    if (tid < 192) { sh_inpz[1][tid] = 0; sh_inpz[0][tid] = 0; }
    float c_reg = 0.0f, acc_reg = 0.0f, cnt = 0.0f;   // live for tid < H_
    __syncthreads();

    const int nt0   = wave;
    const int nt1   = wave + 16;
    const int keep  = ((lane & 15) == 0);   // only A rows M=0 are real
    const int khalf = lane >> 4;
    // lane-constant A-fragment base: real data or the zero region
    const unsigned short* abase = sh_inpz[keep ? 0 : 1] + khalf * 8;

    for (int j = 0; j < T_; ++j) {
        const int   mj    = (int)Xm[j];
        const float tj    = Xt[j];
        const float tnext = (j + 1 < T_) ? Xt[j + 1] : (tj + 1.0f);
        const bool  valid    = (j < len);
        const bool  boundary = (j == len - 1) || (tnext != tj);
        const float dm    = W_dec[mj] * Xd[j] + b_dec[mj];
        const float decay = __expf(-fmaxf(0.0f, dm));

        // ---- stage 1: inp_vec = [x_j, decay*h_t[mj], 0-pad] as bf16 in LDS
        if (tid < KPAD) {
            float v = 0.0f;
            if (tid == 0)        v = Xx[j];
            else if (tid <= H_)  v = decay * sh_h[mj * H_ + (tid - 1)];
            sh_inpz[0][tid] = f2bf(v);
        }
        __syncthreads();

        const unsigned short* wbase = packW + ((size_t)mj * NT * NKC) * 512;

        // warm the NEAR cache (WGP scope: "pull into all levels") with the
        // full next-step slab this wave will read; L2 already holds the pack.
        if (j + 1 < T_) {
            int mn = (int)Xm[j + 1];
            const unsigned short* pn0 =
                packW + (((size_t)mn * NT + nt0) * NKC) * 512 + lane * 16;
            const unsigned short* pn1 =
                packW + (((size_t)mn * NT + nt1) * NKC) * 512 + lane * 16;
            #pragma unroll
            for (int kc = 0; kc < NKC; ++kc) {
                __builtin_prefetch(pn0 + kc * 512, 0, 3);  // -> global_prefetch_b8 (WGP)
                __builtin_prefetch(pn1 + kc * 512, 0, 3);
            }
        }

        // ---- stage 2: GEMV via v_wmma_f32_16x16x32_bf16, row 0 of A = inp_vec
        v8f acc0 = {}, acc1 = {};
        #pragma unroll
        for (int kc = 0; kc < NKC; ++kc) {
            union { v16bf v; v4u q[2]; } a;
            const v4u* ip = (const v4u*)(abase + kc * 32);
            a.q[0] = ip[0];     // elements 0..7  (K base + {0..7}/{8..15})
            a.q[1] = ip[2];     // elements 8..15 (K base + {16..23}/{24..31})

            union { v16bf v; v4u q[2]; } bu0, bu1;
            const v4u* bp0 = (const v4u*)(wbase + ((size_t)(nt0 * NKC + kc)) * 512 + lane * 16);
            const v4u* bp1 = (const v4u*)(wbase + ((size_t)(nt1 * NKC + kc)) * 512 + lane * 16);
            bu0.q[0] = bp0[0]; bu0.q[1] = bp0[1];
            bu1.q[0] = bp1[0]; bu1.q[1] = bp1[1];

            acc0 = __builtin_amdgcn_wmma_f32_16x16x32_bf16(
                       false, a.v, false, bu0.v, (short)0, acc0, false, false);
            acc1 = __builtin_amdgcn_wmma_f32_16x16x32_bf16(
                       false, a.v, false, bu1.v, (short)0, acc1, false, false);
        }
        // D row M=0 sits in VGPR0 (acc[0]) of lanes 0..15, N = lane
        if (lane < 16) {
            sh_gates[nt0 * 16 + lane] = acc0[0] + b_lin[(size_t)mj * H4_ + nt0 * 16 + lane];
            sh_gates[nt1 * 16 + lane] = acc1[0] + b_lin[(size_t)mj * H4_ + nt1 * 16 + lane];
        }
        __syncthreads();

        // ---- stage 3: pointwise LSTM update on H=128 lanes
        if (tid < H_) {
            float gi = sh_gates[tid];
            float gf = sh_gates[H_ + tid];
            float go = sh_gates[2 * H_ + tid];
            float gc = sh_gates[3 * H_ + tid];
            float ccand = sigmoidf_(gf) * c_reg + sigmoidf_(gi) * tanhf(gc);
            float hrow  = sigmoidf_(go) * tanhf(ccand);
            if (valid) {
                sh_h[mj * H_ + tid] = hrow;
                acc_reg += ccand;
                cnt     += 1.0f;
            }
            if (valid && boundary) {
                c_reg   = acc_reg / fmaxf(cnt, 1.0f);
                acc_reg = 0.0f;
                cnt     = 0.0f;
            }
        }
        __syncthreads();
    }

    // ---- output head: feats = [c_t(128), h_t(8192)] @ W_out(8320x2), softmax
    if (tid < H_) sh_gates[tid] = c_reg;
    __syncthreads();
    float p0 = 0.0f, p1 = 0.0f;
    const int NF = F_ * H_ + H_;   // 8320
    for (int i = tid; i < NF; i += 512) {
        float fv = (i < H_) ? sh_gates[i] : sh_h[i - H_];
        p0 += fv * W_out[2 * i];
        p1 += fv * W_out[2 * i + 1];
    }
    sh_r0[tid] = p0; sh_r1[tid] = p1;
    __syncthreads();
    for (int s = 256; s > 0; s >>= 1) {
        if (tid < s) { sh_r0[tid] += sh_r0[tid + s]; sh_r1[tid] += sh_r1[tid + s]; }
        __syncthreads();
    }
    if (tid == 0) {
        float l0 = sh_r0[0] + b_out[0];
        float l1 = sh_r1[0] + b_out[1];
        float m  = fmaxf(l0, l1);
        float e0 = __expf(l0 - m), e1 = __expf(l1 - m);
        float inv = 1.0f / (e0 + e1);
        out[b * 2 + 0] = e0 * inv;
        out[b * 2 + 1] = e1 * inv;
    }
}

extern "C" void kernel_launch(void* const* d_in, const int* in_sizes, int n_in,
                              void* d_out, int out_size, void* d_ws, size_t ws_size,
                              hipStream_t stream) {
    const float* X      = (const float*)d_in[0];
    const int*   lens   = (const int*)  d_in[1];
    const float* W_lin  = (const float*)d_in[2];
    const float* b_lin  = (const float*)d_in[3];
    const float* W_dec  = (const float*)d_in[4];
    const float* b_dec  = (const float*)d_in[5];
    const float* W_out  = (const float*)d_in[6];
    const float* b_out  = (const float*)d_in[7];
    unsigned short* packW = (unsigned short*)d_ws;   // 10.5 MB bf16 weight pack

    const int total = F_ * NT * NKC * 512;
    pack_weights<<<(total + 255) / 256, 256, 0, stream>>>(W_lin, packW);
    seq_scan<<<B_, 512, 0, stream>>>(X, lens, packW, b_lin, W_dec, b_dec,
                                     W_out, b_out, (float*)d_out);
}
